// HTR_50208167690482
// MI455X (gfx1250) — matrix operand order repaired and checked by reading, not compiled, not run on hardware
//
#include <hip/hip_runtime.h>
#include <hip/hip_bf16.h>

// ---------------------------------------------------------------------------
// MI455X (gfx1250) fused kernel for the HTR edge-attention reference.
//   Memory: ~1.7 GB compulsory traffic -> ~73us @ 23.3 TB/s  (HBM-bound)
//   Compute: ~110 GFLOP as v_wmma_f32_16x16x32_f16 (f32 accumulate) ~15-25us
// Rejection folded algebraically:  sum_m qr*kr = S - Pq*Pk*(2 - R)
// X tiles staged cooperatively to LDS as f16 (double-buffered), so the
// f32->f16 conversion happens once per workgroup instead of once per wave.
// ---------------------------------------------------------------------------

typedef __attribute__((ext_vector_type(4)))  float    v4f;
typedef __attribute__((ext_vector_type(8)))  float    v8f;
typedef __attribute__((ext_vector_type(8)))  _Float16 v8h;
typedef __attribute__((ext_vector_type(16))) _Float16 v16h;

#define EPB   16      // edges per workgroup
#define NM    24      // sum of degrees (3+5+7+9)
#define SPH   128
#define HIDW  128
#define EDGEF 128

#define WMMA_F16(acc, A, B) \
  (acc) = __builtin_amdgcn_wmma_f32_16x16x32_f16(false, (A), false, (B), (short)0, (acc), false, false)

__device__ __forceinline__ v8f load8f(const float* __restrict__ p) {
  v4f a = *(const v4f*)p;
  v4f b = *(const v4f*)(p + 4);
  return __builtin_shufflevector(a, b, 0,1,2,3,4,5,6,7);
}
__device__ __forceinline__ v8h cvt8_f32_f16(const float* __restrict__ p) {
  return __builtin_convertvector(load8f(p), v8h);
}
__device__ __forceinline__ v16h cat8(v8h lo, v8h hi) {
  return __builtin_shufflevector(lo, hi, 0,1,2,3,4,5,6,7,8,9,10,11,12,13,14,15);
}
// A operand (16x32 f16), per-lane, from a global f32 row (row = this lane's M)
// lanes 0-15 hold K {k0..k0+7, k0+16..k0+23}; lanes 16-31 the +8 variants.
__device__ __forceinline__ v16h a_frag_g(const float* __restrict__ row, int kc, int hi8) {
  const int k0 = kc * 32 + hi8;                       // hi8 in {0,8}
  return cat8(cvt8_f32_f16(row + k0), cvt8_f32_f16(row + k0 + 16));
}
// B operand (32x16 f16) = W^T tile; column N = hglob, per-lane.
// lanes 0-15: K = kc*32+{0..7,16..23} ; lanes 16-31: K = kc*32+{8..15,24..31}.
__device__ __forceinline__ v16h b_frag_g(const float* __restrict__ W, int hglob, int kc, int hi16) {
  const float* p = W + (long)hglob * 128 + kc * 32 + (hi16 >> 1);  // hi16 in {0,16}
  return cat8(cvt8_f32_f16(p), cvt8_f32_f16(p + 16));
}
// A operand from an f16 LDS row (128 wide): two ds_load_b128
__device__ __forceinline__ v16h a_frag_lds(const _Float16* row, int kc, int hi8) {
  const int k0 = kc * 32 + hi8;
  v8h lo = *(const v8h*)(row + k0);
  v8h hv = *(const v8h*)(row + k0 + 16);
  return cat8(lo, hv);
}
__device__ __forceinline__ float silu_f(float x) {
  return x * (1.0f / (1.0f + __expf(-x)));
}

__global__ __launch_bounds__(256, 1)
void htr_fused_kernel(const float* __restrict__ t_ij,
                      const float* __restrict__ X_i,
                      const float* __restrict__ X_j,
                      const float* __restrict__ rl_ij,
                      const float* __restrict__ W_vq,
                      const float* __restrict__ W_vk,
                      const float* __restrict__ gw_w,
                      const float* __restrict__ gw_b,
                      const float* __restrict__ gt_w1,
                      const float* __restrict__ gt_b1,
                      const float* __restrict__ gt_w2,
                      const float* __restrict__ gt_b2,
                      float* __restrict__ out)
{
  __shared__ __align__(16) _Float16 xibuf[2][EPB * 128];  // staged X_i m-slice (f16)
  __shared__ __align__(16) _Float16 xjbuf[2][EPB * 128];  // staged X_j m-slice (f16)
  __shared__ __align__(16) _Float16 wbuf[EPB * 128];      // w_ij tile (f16)
  __shared__ __align__(16) _Float16 ubuf[EPB * 128];      // hidden MLP tile (f16)
  __shared__ float rlbuf[EPB * NM];
  __shared__ float Rbuf[EPB * 4];

  const int offs[4] = {0, 3, 8, 15};
  const int degs[4] = {3, 5, 7, 9};

  const int e0   = blockIdx.x * EPB;
  const int tid  = threadIdx.x;
  const int wave = tid >> 5;          // 0..7  -> h-tile index
  const int lane = tid & 31;
  const int ht   = wave;
  const int lrow = lane & 15;         // A-row / B-col / C-col for this lane
  const int hi8  = (lane >= 16) ? 8  : 0;
  const int hi16 = (lane >= 16) ? 16 : 0;
  const int cM   = (lane >= 16) ? 8  : 0;   // C fragment row base
  const int hglob = ht * 16 + lrow;

  // staging role: thread -> (edge row, 8-float column chunk), fully coalesced
  const int srow = tid >> 4;           // 0..15
  const int scol = (tid & 15) * 8;     // 0..120
  const float* XiS = X_i + ((long)(e0 + srow) * NM) * SPH + scol;
  const float* XjS = X_j + ((long)(e0 + srow) * NM) * SPH + scol;

  // ---- stage rl tile + per-(edge,l) R = sum rl^2 into LDS -----------------
  for (int i = tid; i < EPB * NM; i += 256)
    rlbuf[i] = rl_ij[(long)(e0 + i / NM) * NM + (i % NM)];
  __syncthreads();
  if (tid < EPB * 4) {
    const int e = tid >> 2, l = tid & 3;
    const int s = offs[l], d = degs[l];
    float r = 0.0f;
    for (int m = 0; m < d; ++m) { float v = rlbuf[e * NM + s + m]; r += v * v; }
    Rbuf[tid] = r;
  }

  // ---- weight B fragments (Wq constant across l) --------------------------
  v16h Bq[4], Bk[4];
  #pragma unroll
  for (int kc = 0; kc < 4; ++kc) {
    Bq[kc] = b_frag_g(W_vq, hglob, kc, hi16);
    Bk[kc] = b_frag_g(W_vk /* l = 0 */, hglob, kc, hi16);
  }

  // ---- prologue: stage slice g = 0 ----------------------------------------
  {
    v8h si = __builtin_convertvector(load8f(XiS), v8h);
    v8h sj = __builtin_convertvector(load8f(XjS), v8h);
    *(v8h*)(&xibuf[0][srow * 128 + scol]) = si;
    *(v8h*)(&xjbuf[0][srow * 128 + scol]) = sj;
  }
  __syncthreads();

  // ---- einsum + rejection contraction (flattened over g = (l,m)) ----------
  const v8f vzero = {};
  v8f w_acc = vzero, S = vzero, Pq = vzero, Pk = vzero;
  int l = 0;

  for (int g = 0; g < NM; ++g) {
    const int buf = g & 1;

    // issue next slice's global loads before the WMMAs (latency overlap)
    v8f vi = vzero, vj = vzero;
    if (g + 1 < NM) {
      vi = load8f(XiS + (long)(g + 1) * SPH);
      vj = load8f(XjS + (long)(g + 1) * SPH);
    }
    if (g + 2 < NM) {
      __builtin_prefetch(XiS + (long)(g + 2) * SPH, 0, 3);  // global_prefetch_b8
      __builtin_prefetch(XjS + (long)(g + 2) * SPH, 0, 3);
    }

    // q/k for this m-slice: A from LDS (ds_load_b128), B from registers
    v8f q = vzero, k = vzero;
    const _Float16* xri = &xibuf[buf][lrow * 128];
    const _Float16* xrj = &xjbuf[buf][lrow * 128];
    #pragma unroll
    for (int kc = 0; kc < 4; ++kc) {
      v16h Ai = a_frag_lds(xri, kc, hi8);
      WMMA_F16(q, Ai, Bq[kc]);
    }
    #pragma unroll
    for (int kc = 0; kc < 4; ++kc) {
      v16h Aj = a_frag_lds(xrj, kc, hi8);
      WMMA_F16(k, Aj, Bk[kc]);
    }
    #pragma unroll
    for (int v = 0; v < 8; ++v) {
      const float rlv = rlbuf[(cM + v) * NM + g];   // rl for C-row M=cM+v
      S[v]  += q[v] * k[v];
      Pq[v] += rlv * q[v];
      Pk[v] += rlv * k[v];
    }

    // end of an l-block? fold S/Pq/Pk into w_acc (uniform branch)
    if (g == offs[l] + degs[l] - 1) {
      const float invd = 1.0f / (float)degs[l];
      #pragma unroll
      for (int v = 0; v < 8; ++v) {
        const float Rv = Rbuf[(cM + v) * 4 + l];
        w_acc[v] += (S[v] - Pq[v] * Pk[v] * (2.0f - Rv)) * invd;
      }
      S = vzero; Pq = vzero; Pk = vzero;
      ++l;
      if (l < 4) {
        #pragma unroll
        for (int kc = 0; kc < 4; ++kc)
          Bk[kc] = b_frag_g(W_vk + (long)l * HIDW * SPH, hglob, kc, hi16);
      }
    }

    // commit next slice to the other buffer, then one barrier per step
    if (g + 1 < NM) {
      *(v8h*)(&xibuf[buf ^ 1][srow * 128 + scol]) = __builtin_convertvector(vi, v8h);
      *(v8h*)(&xjbuf[buf ^ 1][srow * 128 + scol]) = __builtin_convertvector(vj, v8h);
      __syncthreads();
    }
  }

  // publish w tile (f16) for the gw GEMM
  #pragma unroll
  for (int v = 0; v < 8; ++v)
    wbuf[(cM + v) * 128 + hglob] = (_Float16)w_acc[v];
  __syncthreads();

  // ---- MLP 1: u = silu(t @ gt_w1^T + b1)  (A from global t rows) ----------
  const float* trow = t_ij + (long)(e0 + lrow) * EDGEF;
  v8f u = vzero;
  {
    v16h B1[4];
    #pragma unroll
    for (int kc = 0; kc < 4; ++kc) B1[kc] = b_frag_g(gt_w1, hglob, kc, hi16);
    #pragma unroll
    for (int kc = 0; kc < 4; ++kc) {
      v16h At = a_frag_g(trow, kc, hi8);
      WMMA_F16(u, At, B1[kc]);
    }
    const float b1v = gt_b1[hglob];
    #pragma unroll
    for (int v = 0; v < 8; ++v) {
      u[v] = silu_f(u[v] + b1v);
      ubuf[(cM + v) * 128 + hglob] = (_Float16)u[v];
    }
  }

  // ---- MLP gw = silu(w @ gw_w^T + gw_b)  (A from LDS wbuf, ready above) ---
  v8f gw = vzero;
  {
    v16h Bg[4];
    #pragma unroll
    for (int kc = 0; kc < 4; ++kc) Bg[kc] = b_frag_g(gw_w, hglob, kc, hi16);
    const _Float16* wrow = wbuf + lrow * 128;
    #pragma unroll
    for (int kc = 0; kc < 4; ++kc) {
      v16h Aw = a_frag_lds(wrow, kc, hi8);
      WMMA_F16(gw, Aw, Bg[kc]);
    }
    const float gbv = gw_b[hglob];
    #pragma unroll
    for (int v = 0; v < 8; ++v) gw[v] = silu_f(gw[v] + gbv);
  }
  __syncthreads();   // ubuf complete for all waves

  // ---- MLP 2: gt = silu(u @ gt_w2^T + b2) ---------------------------------
  v8f gt = vzero;
  {
    v16h B2[4];
    #pragma unroll
    for (int kc = 0; kc < 4; ++kc) B2[kc] = b_frag_g(gt_w2, hglob, kc, hi16);
    const _Float16* urow = ubuf + lrow * 128;
    #pragma unroll
    for (int kc = 0; kc < 4; ++kc) {
      v16h Au = a_frag_lds(urow, kc, hi8);
      WMMA_F16(gt, Au, B2[kc]);
    }
    const float b2v = gt_b2[hglob];
    #pragma unroll
    for (int v = 0; v < 8; ++v) gt[v] = silu_f(gt[v] + b2v);
  }

  // ---- out = t + gw * gt --------------------------------------------------
  #pragma unroll
  for (int v = 0; v < 8; ++v) {
    const long idx = (long)(e0 + cM + v) * EDGEF + hglob;
    out[idx] = t_ij[idx] + gw[v] * gt[v];
  }
}

extern "C" void kernel_launch(void* const* d_in, const int* in_sizes, int n_in,
                              void* d_out, int out_size, void* d_ws, size_t ws_size,
                              hipStream_t stream) {
  (void)n_in; (void)d_ws; (void)ws_size; (void)out_size;
  const float* t_ij  = (const float*)d_in[0];
  const float* X_i   = (const float*)d_in[1];
  const float* X_j   = (const float*)d_in[2];
  const float* rl_ij = (const float*)d_in[3];
  const float* W_vq  = (const float*)d_in[4];
  const float* W_vk  = (const float*)d_in[5];
  const float* gw_w  = (const float*)d_in[6];
  const float* gw_b  = (const float*)d_in[7];
  const float* gt_w1 = (const float*)d_in[8];
  const float* gt_b1 = (const float*)d_in[9];
  const float* gt_w2 = (const float*)d_in[10];
  const float* gt_b2 = (const float*)d_in[11];
  float* out = (float*)d_out;

  const int E = in_sizes[0] / EDGEF;      // 65536
  const int blocks = E / EPB;             // 4096
  htr_fused_kernel<<<blocks, 256, 0, stream>>>(
      t_ij, X_i, X_j, rl_ij, W_vq, W_vk,
      gw_w, gw_b, gt_w1, gt_b1, gt_w2, gt_b2, out);
}
